// RayTransformer_3728031612936
// MI455X (gfx1250) — matrix operand order repaired and checked by reading, not compiled, not run on hardware
//
// MI455X (gfx1250) RayTransformer pipeline.
// - f16 WMMA operands (v_wmma_f32_16x16x32_f16), f32 accumulate, f32 scalar
//   reductions / layernorm / softmax. wave32; dynamic LDS staging for all GEMMs.
// - rgb_feat is stored f16 in workspace and pulled into the WMMA A-tile with
//   GLOBAL_LOAD_ASYNC_TO_LDS_B32 (ASYNCcnt), overlapping the blend-weight math.
// - global_prefetch_b8 warms bilinear tap cachelines in the gather kernel.
// Workspace requirement: ~16.5M floats (~66 MB).

#include <hip/hip_runtime.h>

typedef __attribute__((ext_vector_type(16))) _Float16 v16h;
typedef __attribute__((ext_vector_type(8)))  float    v8f;

#define ACT_NONE 0
#define ACT_ELU  1
#define ACT_RELU 2

constexpr int BB = 2, NVV = 4, RNN = 1024, SNN = 64, IH = 256, IW = 256, FDIM = 32;
constexpr int NPT = BB * NVV * RNN * SNN;   // 524288 per-view points
constexpr int NPS = BB * RNN * SNN;         // 131072 ray samples

// workspace layout (float offsets)
constexpr size_t OFF_RF    = 0;                                // rgb_feat f16 [NPT][36] (==NPT*18 floats)
constexpr size_t OFF_RD    = OFF_RF    + (size_t)NPT * 18;     // ray_diff  [NPT][4]
constexpr size_t OFF_RGBIN = OFF_RD    + (size_t)NPT * 4;      // rgb_in    [NPT][3]
constexpr size_t OFF_MASK  = OFF_RGBIN + (size_t)NPT * 3;      // mask      [NPT]
constexpr size_t OFF_LOGIT = OFF_MASK  + (size_t)NPT;          // blend logit [NPT]
constexpr size_t OFF_GF    = OFF_LOGIT + (size_t)NPT;          // geo feat  [NPS][16]
constexpr size_t OFF_NUMV  = OFF_GF    + (size_t)NPS * 16;     // num_valid [NPS]
constexpr size_t OFF_SIGMA = OFF_NUMV  + (size_t)NPS;          // sigma     [NPS]
constexpr size_t OFF_LMAX  = OFF_SIGMA + (size_t)NPS;          // softmax max [B*NV*SN]
constexpr size_t OFF_LSUM  = OFF_LMAX  + (size_t)BB * NVV * SNN;

constexpr int SMEM_B = 153120;   // dynamic LDS for per-ray MLP kernel
constexpr int SMEM_C = 72064;    // dynamic LDS for attention kernel

struct Params {
  const float *rd0w, *rd0b, *rd1w, *rd1b;
  const float *base0w, *base0b, *base1w, *base1b;
  const float *vis0w, *vis0b, *vis1w, *vis1b;
  const float *v20w, *v20b, *v21w, *v21b;
  const float *geo0w, *geo0b, *geo1w, *geo1b;
  const float *og0w, *og0b, *og1w, *og1b;
  const float *rgb0w, *rgb0b, *rgb1w, *rgb1b, *rgb2w, *rgb2b;
  const float *aq, *ak, *av, *amerge, *amlp0, *amlp1;
  const float *ln1w, *ln1b, *ln2w, *ln2b;
  const float *sparam;
};

__device__ __forceinline__ float eluf(float x) { return x > 0.f ? x : __expf(x) - 1.f; }
__device__ __forceinline__ float act_apply(float x, int act) {
  if (act == ACT_ELU)  return eluf(x);
  if (act == ACT_RELU) return fmaxf(x, 0.f);
  return x;
}

// ---- WMMA fragment loaders (16-bit A 16x32 / B 32x16 layouts per CDNA5 ISA) ----
__device__ __forceinline__ v16h frag_a(const _Float16* A, int lda, int lane) {
  const int m = lane & 15, g = lane >> 4;
  const _Float16* p = A + m * lda + 8 * g;       // K = 8g..8g+7 and 16+8g..16+8g+7
  v16h r;
#pragma unroll
  for (int i = 0; i < 8; ++i) r[i] = p[i];
#pragma unroll
  for (int i = 0; i < 8; ++i) r[8 + i] = p[16 + i];
  return r;
}
__device__ __forceinline__ v16h frag_b(const _Float16* Wm, int ldw, int lane) {
  const int n = lane & 15, g = lane >> 4;        // K = 8g..8g+7 and 16+8g..16+8g+7, N = n
  const _Float16* p = Wm + n * ldw + 8 * g;      // weights stored row-major [out][in]
  v16h r;
#pragma unroll
  for (int i = 0; i < 8; ++i) r[i] = p[i];
#pragma unroll
  for (int i = 0; i < 8; ++i) r[8 + i] = p[16 + i];
  return r;
}

// OUT(MxNp) = act(A(MxKp f16,row-major lda) * W(NpxKp f16,row-major ldw)^T + bias)
// Job-parallel over 16x16 (mt,nt) tiles across all waves of the block.
__device__ void gemm_wmma(const _Float16* A, int lda, const _Float16* Wm, int ldw,
                          const float* bias, _Float16* out16, int ldo16,
                          float* out32, int ldo32, int M, int Np, int Kp, int act) {
  const int lane = threadIdx.x & 31;
  const int wave = threadIdx.x >> 5;
  const int nwaves = blockDim.x >> 5;
  const int nmt = M >> 4, nnt = Np >> 4, nkt = Kp >> 5;
  for (int job = wave; job < nmt * nnt; job += nwaves) {
    const int mt = job / nnt, nt = job % nnt;
    v8f acc = {};
    for (int kt = 0; kt < nkt; ++kt) {
      v16h a = frag_a(A + mt * 16 * lda + kt * 32, lda, lane);
      v16h b = frag_b(Wm + nt * 16 * ldw + kt * 32, ldw, lane);
      acc = __builtin_amdgcn_wmma_f32_16x16x32_f16(false, a, false, b, (short)0, acc,
                                                   false, false);
    }
    const int n = lane & 15, g = lane >> 4;
    const int col = nt * 16 + n;
    const float bv = bias ? bias[col] : 0.f;
#pragma unroll
    for (int r = 0; r < 8; ++r) {
      const int row = mt * 16 + r + 8 * g;
      float v = act_apply(acc[r] + bv, act);
      if (out16) out16[row * ldo16 + col] = (_Float16)v;
      if (out32) out32[row * ldo32 + col] = v;
    }
  }
}

// Stage global f32 weight (Nreal x Kreal) into LDS f16 padded (Np x Kp) + padded bias.
__device__ void stage_w(const float* Wg, int Nreal, int Kreal, const float* bg,
                        _Float16* Wl, float* bl, int Np, int Kp) {
  for (int i = threadIdx.x; i < Np * Kp; i += blockDim.x) {
    int nn = i / Kp, kk = i % Kp;
    Wl[i] = (_Float16)((nn < Nreal && kk < Kreal) ? Wg[nn * Kreal + kk] : 0.f);
  }
  for (int i = threadIdx.x; i < Np; i += blockDim.x)
    bl[i] = (bg && i < Nreal) ? bg[i] : 0.f;
}

// ============ Kernel A: projection, bilinear gather, ray-dir MLP ============
__global__ __launch_bounds__(256)
void k_project(const float* __restrict__ p3d, const float* __restrict__ simgs,
               const float* __restrict__ rpi, const float* __restrict__ spi,
               const float* __restrict__ spose, const float* __restrict__ sfeat,
               Params P, float* __restrict__ ws, float* __restrict__ out) {
  __shared__ float sw0[64], sb0[16], sw1[560], sb1[35];
  for (int i = threadIdx.x; i < 64; i += 256) sw0[i] = P.rd0w[i];
  for (int i = threadIdx.x; i < 16; i += 256) sb0[i] = P.rd0b[i];
  for (int i = threadIdx.x; i < 560; i += 256) sw1[i] = P.rd1w[i];
  for (int i = threadIdx.x; i < 35; i += 256) sb1[i] = P.rd1b[i];
  __syncthreads();

  const int idx = blockIdx.x * 256 + threadIdx.x;     // = ((b*NV+v)*RN+r)*SN+s
  const int s = idx % SNN;
  const int r = (idx / SNN) % RNN;
  const int v = (idx / (SNN * RNN)) % NVV;
  const int b = idx / (SNN * RNN * NVV);

  const size_t pbase = ((size_t)(b * RNN + r) * SNN + s) * 3;
  const float p0 = p3d[pbase], p1 = p3d[pbase + 1], p2 = p3d[pbase + 2];

  const float* rp = rpi + b * 16;
  float a0 = p0 - rp[3], a1 = p1 - rp[7], a2 = p2 - rp[11];
  float inv = rsqrtf(a0 * a0 + a1 * a1 + a2 * a2);
  a0 *= inv; a1 *= inv; a2 *= inv;

  const float* sp = spi + (b * NVV + v) * 16;
  float c0 = p0 - sp[3], c1 = p1 - sp[7], c2 = p2 - sp[11];
  inv = rsqrtf(c0 * c0 + c1 * c1 + c2 * c2);
  c0 *= inv; c1 *= inv; c2 *= inv;

  const float dotv = a0 * c0 + a1 * c1 + a2 * c2;
  float rd[4] = {a0 - c0, a1 - c1, a2 - c2, dotv};

  const float* PM = spose + (b * NVV + v) * 16;
  const float X = PM[0] * p0 + PM[1] * p1 + PM[2] * p2 + PM[3];
  const float Y = PM[4] * p0 + PM[5] * p1 + PM[6] * p2 + PM[7];
  const float Zc = PM[8] * p0 + PM[9] * p1 + PM[10] * p2 + PM[11];
  const float mvd = Zc > 0.f ? 1.f : 0.f;
  const float gx = X / Zc, gy = Y / Zc;

  out[(size_t)NPS * 4 + (size_t)idx * 2 + 0] = gx;   // points_in_pixel
  out[(size_t)NPS * 4 + (size_t)idx * 2 + 1] = gy;

  const float inb = (gx >= 0.f && gx <= (float)(IW - 1) &&
                     gy >= 0.f && gy <= (float)(IH - 1)) ? 1.f : 0.f;
  const float fx0 = floorf(gx), fy0 = floorf(gy);
  const float wx = gx - fx0, wy = gy - fy0;
  const float txs[4] = {fx0, fx0 + 1.f, fx0, fx0 + 1.f};
  const float tys[4] = {fy0, fy0, fy0 + 1.f, fy0 + 1.f};
  const float tws[4] = {(1.f - wx) * (1.f - wy), wx * (1.f - wy),
                        (1.f - wx) * wy, wx * wy};
  int off[4]; float wt[4];
#pragma unroll
  for (int t = 0; t < 4; ++t) {
    const float xi = txs[t], yi = tys[t];
    const float valid = (xi >= 0.f && xi < (float)IW && yi >= 0.f && yi < (float)IH)
                        ? 1.f : 0.f;
    const int xc = (int)fminf(fmaxf(xi, 0.f), (float)(IW - 1));
    const int yc = (int)fminf(fmaxf(yi, 0.f), (float)(IH - 1));
    off[t] = yc * IW + xc;
    wt[t] = tws[t] * valid;
  }
  const size_t HW = (size_t)IH * IW;
  const float* ib = simgs + (size_t)(b * NVV + v) * 3 * HW;
  const float* fb = sfeat + (size_t)(b * NVV + v) * FDIM * HW;

  // warm the gather cachelines (taps 0/1 and 2/3 usually share a line)
  for (int c = 0; c < FDIM; ++c) {
    const float* pc = fb + (size_t)c * HW;
    __builtin_prefetch(pc + off[0], 0, 1);
    __builtin_prefetch(pc + off[2], 0, 1);
  }

  float rgb[3];
#pragma unroll
  for (int c = 0; c < 3; ++c) {
    const float* pc = ib + c * HW;
    rgb[c] = wt[0] * pc[off[0]] + wt[1] * pc[off[1]] +
             wt[2] * pc[off[2]] + wt[3] * pc[off[3]];
  }

  // ray-dir MLP: 4 -> 16 -> 35 (elu)
  float h1[16];
#pragma unroll
  for (int i = 0; i < 16; ++i) {
    float a = sb0[i];
#pragma unroll
    for (int j = 0; j < 4; ++j) a += sw0[i * 4 + j] * rd[j];
    h1[i] = eluf(a);
  }

  const float mask = inb * mvd;
  const size_t g = (size_t)idx;
  ws[OFF_MASK + g] = mask;
#pragma unroll
  for (int j = 0; j < 4; ++j) ws[OFF_RD + g * 4 + j] = rd[j];
#pragma unroll
  for (int c = 0; c < 3; ++c) ws[OFF_RGBIN + g * 3 + c] = rgb[c];

  _Float16* rf16 = (_Float16*)(ws + OFF_RF);        // [NPT][36] halves
  for (int c = 0; c < 35; ++c) {
    float d = sb1[c];
#pragma unroll
    for (int j = 0; j < 16; ++j) d += sw1[c * 16 + j] * h1[j];
    d = eluf(d);
    float samp;
    if (c < 32) {
      const float* pc = fb + (size_t)c * HW;
      samp = wt[0] * pc[off[0]] + wt[1] * pc[off[1]] +
             wt[2] * pc[off[2]] + wt[3] * pc[off[3]];
    } else {
      samp = rgb[c - 32];
    }
    rf16[g * 36 + c] = (_Float16)(samp + d);
  }
  rf16[g * 36 + 35] = (_Float16)0.f;                // pad half (lands in A16 col 105)
}

// ============ Kernel B: per-ray WMMA MLP chain (1 WG = 1 ray, 256 pts) ============
__global__ __launch_bounds__(256)
void k_perray(Params P, float* __restrict__ ws) {
  extern __shared__ unsigned char smem[];
  _Float16* A16 = (_Float16*)smem;              // 256*128 halves  (at LDS offset 0!)
  _Float16* B16 = A16 + 256 * 128;              // 256*64
  _Float16* W16 = B16 + 256 * 64;               // 64*128
  float* x32    = (float*)(W16 + 64 * 128);     // 256*32
  float* biasL  = x32 + 256 * 32;               // 64
  float* wbuf   = biasL + 64;                   // 256
  float* w2buf  = wbuf + 256;                   // 256
  float* visbuf = w2buf + 256;                  // 256
  float* maskb  = visbuf + 256;                 // 256
  float* red    = maskb + 256;                  // 256
  float* sarr   = red + 256;                    // 64
  float* vsum   = sarr + 64;                    // 8

  const int tid = threadIdx.x;
  const int b = blockIdx.x / RNN, r = blockIdx.x % RNN;
  const int v = tid >> 6, s = tid & 63;
  const size_t gidx = ((size_t)(b * NVV + v) * RNN + r) * SNN + s;

  // Async-copy this point's 36 f16 rgb_feat values straight into the WMMA A-tile:
  // A16[tid][70..105] lives at dynamic-LDS byte offset tid*256+140 (A16 is the
  // first carve, no static LDS in this kernel). ASYNCcnt tracks completion; the
  // copy overlaps the blend-weight reductions below.
  {
    const unsigned lds0 = (unsigned)(tid * 256 + 140);
    const unsigned vo0  = (unsigned)(gidx * 72u);
    const unsigned long long sb =
        (unsigned long long)(uintptr_t)(const _Float16*)(ws + OFF_RF);
#pragma unroll
    for (int j = 0; j < 18; ++j) {
      asm volatile("global_load_async_to_lds_b32 %0, %1, %2"
                   :: "v"(lds0 + 4u * j), "v"(vo0 + 4u * j), "s"(sb)
                   : "memory");
    }
  }

  const float mask = ws[OFF_MASK + gidx];
  maskb[tid] = mask;
  const float dotv = ws[OFF_RD + gidx * 4 + 3];
  const float sA = fabsf(P.sparam[0]);
  const float ed = __expf(sA * (dotv - 1.f));
  red[tid] = ed;
  __syncthreads();
  if (tid < 64) {
    float m = red[tid];
    for (int vv = 1; vv < NVV; ++vv) m = fminf(m, red[vv * 64 + tid]);
    sarr[tid] = m;
  }
  __syncthreads();
  const float wraw = (ed - sarr[s]) * mask;
  red[tid] = wraw;
  __syncthreads();
  if (tid < 64) {
    float m = 0.f;
    for (int vv = 0; vv < NVV; ++vv) m += red[vv * 64 + tid];
    sarr[tid] = m;
  }
  __syncthreads();
  wbuf[tid] = wraw / (sarr[s] + 1e-8f);

  // zero pad cols 106..127 (col 105 arrives as the async pad half)
  for (int c = 106; c < 128; ++c) A16[tid * 128 + c] = (_Float16)0.f;
  asm volatile("s_wait_asynccnt 0x0" ::: "memory");   // rgb_feat resident in A16
  __syncthreads();

  // weighted mean/var over views -> A16 cols 0..69
  for (int job = tid; job < 64 * 35; job += 256) {
    const int ss = job / 35, c = job % 35;
    float xs[NVV];
    for (int vv = 0; vv < NVV; ++vv) xs[vv] = (float)A16[(vv * 64 + ss) * 128 + 70 + c];
    float m = 0.f;
    for (int vv = 0; vv < NVV; ++vv) m += wbuf[vv * 64 + ss] * xs[vv];
    float var = 0.f;
    for (int vv = 0; vv < NVV; ++vv) {
      const float d = xs[vv] - m;
      var += wbuf[vv * 64 + ss] * d * d;
    }
    for (int vv = 0; vv < NVV; ++vv) {
      A16[(vv * 64 + ss) * 128 + c] = (_Float16)m;
      A16[(vv * 64 + ss) * 128 + 35 + c] = (_Float16)var;
    }
  }
  __syncthreads();

  // base MLP: 105->64 (elu) then 64->32 (elu, f32 kept)
  stage_w(P.base0w, 64, 105, P.base0b, W16, biasL, 64, 128); __syncthreads();
  gemm_wmma(A16, 128, W16, 128, biasL, B16, 64, nullptr, 0, 256, 64, 128, ACT_ELU);
  __syncthreads();
  stage_w(P.base1w, 32, 64, P.base1b, W16, biasL, 32, 64); __syncthreads();
  gemm_wmma(B16, 64, W16, 64, biasL, nullptr, 0, x32, 32, 256, 32, 64, ACT_ELU);
  __syncthreads();

  // vis chain: elu(vis0(x*w)) -> elu(vis1) -> split x_res | vis
  for (int c = 0; c < 32; ++c)
    A16[tid * 32 + c] = (_Float16)(x32[tid * 32 + c] * wbuf[tid]);
  __syncthreads();
  stage_w(P.vis0w, 32, 32, P.vis0b, W16, biasL, 32, 32); __syncthreads();
  gemm_wmma(A16, 32, W16, 32, biasL, B16, 32, nullptr, 0, 256, 32, 32, ACT_ELU);
  __syncthreads();
  stage_w(P.vis1w, 33, 32, P.vis1b, W16, biasL, 48, 32); __syncthreads();
  gemm_wmma(B16, 32, W16, 32, biasL, A16, 48, nullptr, 0, 256, 48, 32, ACT_ELU);
  __syncthreads();
  {
    float xr[32];
    for (int c = 0; c < 32; ++c) xr[c] = (float)A16[tid * 48 + c];
    const float vl = (float)A16[tid * 48 + 32];
    const float visv = (1.f / (1.f + __expf(-vl))) * mask;
    __syncthreads();   // done reading A16 (ld48) before rewriting (ld32)
    for (int c = 0; c < 32; ++c) x32[tid * 32 + c] += xr[c];
    visbuf[tid] = visv;
    for (int c = 0; c < 32; ++c)
      A16[tid * 32 + c] = (_Float16)(x32[tid * 32 + c] * visv);
  }
  __syncthreads();

  // vis2 chain
  stage_w(P.v20w, 32, 32, P.v20b, W16, biasL, 32, 32); __syncthreads();
  gemm_wmma(A16, 32, W16, 32, biasL, B16, 32, nullptr, 0, 256, 32, 32, ACT_ELU);
  __syncthreads();
  {
    float acc = P.v21b[0];
    for (int c = 0; c < 32; ++c) acc += (float)B16[tid * 32 + c] * P.v21w[c];
    const float visn = (1.f / (1.f + __expf(-acc))) * mask;
    visbuf[tid] = visn;
    red[tid] = visn;
  }
  __syncthreads();
  if (tid < NVV) {
    float sum = 0.f;
    for (int ss = 0; ss < 64; ++ss) sum += red[tid * 64 + ss];
    vsum[tid] = sum;
  }
  __syncthreads();
  w2buf[tid] = visbuf[tid] / (vsum[v] + 1e-8f);
  __syncthreads();

  if (tid < 64) {
    float nvv = 0.f;
    for (int vv = 0; vv < NVV; ++vv) nvv += maskb[vv * 64 + tid];
    ws[OFF_NUMV + (size_t)(b * RNN + r) * SNN + tid] = nvv;
  }
  // geo input tokens: A16[64][96] = [mean2(32) | var2(32) | w2mean | pad]
  for (int job = tid; job < 64 * 96; job += 256) {
    const int ss = job / 96, c = job % 96;
    float val = 0.f;
    if (c < 32) {
      for (int vv = 0; vv < NVV; ++vv)
        val += w2buf[vv * 64 + ss] * x32[(vv * 64 + ss) * 32 + c];
    } else if (c < 64) {
      const int cc = c - 32;
      float m = 0.f;
      for (int vv = 0; vv < NVV; ++vv)
        m += w2buf[vv * 64 + ss] * x32[(vv * 64 + ss) * 32 + cc];
      for (int vv = 0; vv < NVV; ++vv) {
        const float d = x32[(vv * 64 + ss) * 32 + cc] - m;
        val += w2buf[vv * 64 + ss] * d * d;
      }
    } else if (c == 64) {
      for (int vv = 0; vv < NVV; ++vv) val += 0.25f * w2buf[vv * 64 + ss];
    }
    A16[job] = (_Float16)val;
  }
  __syncthreads();

  // geo MLP: 65->64 (elu) -> 64->16 (elu) -> ws gf
  stage_w(P.geo0w, 64, 65, P.geo0b, W16, biasL, 64, 96); __syncthreads();
  gemm_wmma(A16, 96, W16, 96, biasL, B16, 64, nullptr, 0, 64, 64, 96, ACT_ELU);
  __syncthreads();
  stage_w(P.geo1w, 16, 64, P.geo1b, W16, biasL, 16, 64); __syncthreads();
  gemm_wmma(B16, 64, W16, 64, biasL, A16, 16, nullptr, 0, 64, 16, 64, ACT_ELU);
  __syncthreads();
  for (int i = tid; i < 64 * 16; i += 256)
    ws[OFF_GF + (size_t)(b * RNN + r) * SNN * 16 + i] = (float)A16[i];
  __syncthreads();

  // rgb branch: [x(32) | vis | ray_diff(4) | pad] -> 16 (elu) -> scalar 8 -> 1
  for (int c = 0; c < 32; ++c) A16[tid * 64 + c] = (_Float16)x32[tid * 32 + c];
  A16[tid * 64 + 32] = (_Float16)visbuf[tid];
  for (int j = 0; j < 4; ++j)
    A16[tid * 64 + 33 + j] = (_Float16)ws[OFF_RD + gidx * 4 + j];
  for (int c = 37; c < 64; ++c) A16[tid * 64 + c] = (_Float16)0.f;
  __syncthreads();
  stage_w(P.rgb0w, 16, 37, P.rgb0b, W16, biasL, 16, 64); __syncthreads();
  gemm_wmma(A16, 64, W16, 64, biasL, B16, 16, nullptr, 0, 256, 16, 64, ACT_ELU);
  __syncthreads();
  {
    float h8[8];
#pragma unroll
    for (int i = 0; i < 8; ++i) {
      float a = P.rgb1b[i];
      for (int j = 0; j < 16; ++j) a += (float)B16[tid * 16 + j] * P.rgb1w[i * 16 + j];
      h8[i] = eluf(a);
    }
    float a = P.rgb2b[0];
#pragma unroll
    for (int i = 0; i < 8; ++i) a += h8[i] * P.rgb2w[i];
    ws[OFF_LOGIT + gidx] = (mask == 0.f) ? -1000000000.f : a;
  }
}

// ============ Kernel C: linear-attention encoder + out_geo (1 WG = 1 sequence) ====
__global__ __launch_bounds__(128)
void k_attn(Params P, float* __restrict__ ws) {
  extern __shared__ unsigned char smem[];
  _Float16* x16 = (_Float16*)smem;          // 64*32
  _Float16* h16 = x16 + 64 * 32;            // 64*64
  _Float16* o16 = h16 + 64 * 64;            // 64*64
  _Float16* Wc  = o16 + 64 * 64;            // 64*64
  float* x032  = (float*)(Wc + 64 * 64);    // 64*32
  float* qkv   = x032 + 64 * 32;            // 64*96
  float* t32   = qkv + 64 * 96;             // 64*32
  float* KV    = t32 + 64 * 32;             // 256
  float* Ksum  = KV + 256;                  // 32
  float* Zb    = Ksum + 32;                 // 256
  float* biasC = Zb + 256;                  // 64

  const int tid = threadIdx.x;
  const int seq = blockIdx.x;               // b*RN + r

  // x0 = [geo_feat(16) | pos-enc(16)]
  for (int i = tid; i < 64 * 32; i += 128) {
    const int ss = i / 32, c = i % 32;
    float val;
    if (c < 16) {
      val = ws[OFF_GF + (size_t)seq * SNN * 16 + ss * 16 + c];
    } else {
      const int j = c - 16;
      const float e = (float)(2 * (j / 2));
      const float a = (float)ss / powf(10000.f, e / 16.f);
      val = (j & 1) ? cosf(a) : sinf(a);
    }
    x032[i] = val;
    x16[i] = (_Float16)val;
  }
  // fused q|k|v weights (N = 96)
  for (int i = tid; i < 96 * 32; i += 128) {
    const int n = i / 32, k = i % 32;
    const float wv = n < 32 ? P.aq[n * 32 + k]
                    : n < 64 ? P.ak[(n - 32) * 32 + k]
                             : P.av[(n - 64) * 32 + k];
    Wc[i] = (_Float16)wv;
  }
  __syncthreads();
  gemm_wmma(x16, 32, Wc, 32, nullptr, nullptr, 0, qkv, 96, 64, 96, 32, ACT_NONE);
  __syncthreads();
  for (int i = tid; i < 64 * 64; i += 128) {   // Q,K = elu(.)+1
    const int ss = i / 64, c = i % 64;
    qkv[ss * 96 + c] = eluf(qkv[ss * 96 + c]) + 1.f;
  }
  __syncthreads();
  for (int job = tid; job < 256; job += 128) { // KV[h][d][e] = sum_s K*v/L
    const int hh = job >> 6, de = job & 63, d = de >> 3, e = de & 7;
    float a = 0.f;
    for (int ss = 0; ss < 64; ++ss)
      a += qkv[ss * 96 + 32 + hh * 8 + d] * qkv[ss * 96 + 64 + hh * 8 + e];
    KV[job] = a * (1.f / 64.f);
  }
  for (int j = tid; j < 32; j += 128) {        // Ksum
    float a = 0.f;
    for (int ss = 0; ss < 64; ++ss) a += qkv[ss * 96 + 32 + j];
    Ksum[j] = a;
  }
  __syncthreads();
  for (int job = tid; job < 256; job += 128) { // Z
    const int ss = job >> 2, hh = job & 3;
    float a = 0.f;
    for (int d = 0; d < 8; ++d) a += qkv[ss * 96 + hh * 8 + d] * Ksum[hh * 8 + d];
    Zb[job] = 1.f / (a + 1e-6f);
  }
  __syncthreads();
  for (int i = tid; i < 64 * 32; i += 128) {   // msg (f16, staged in h16 ld32)
    const int ss = i / 32, j = i % 32, hh = j >> 3, e = j & 7;
    float a = 0.f;
    for (int d = 0; d < 8; ++d) a += qkv[ss * 96 + hh * 8 + d] * KV[hh * 64 + d * 8 + e];
    h16[i] = (_Float16)(a * Zb[ss * 4 + hh] * 64.f);
  }
  for (int i = tid; i < 32 * 32; i += 128) Wc[i] = (_Float16)P.amerge[i];
  __syncthreads();
  gemm_wmma(h16, 32, Wc, 32, nullptr, nullptr, 0, t32, 32, 64, 32, 32, ACT_NONE);
  __syncthreads();
  if (tid < 64) {                              // ln1 + build h = [x | ln(msg)]
    float m = 0.f;
    for (int c = 0; c < 32; ++c) m += t32[tid * 32 + c];
    m *= (1.f / 32.f);
    float var = 0.f;
    for (int c = 0; c < 32; ++c) { const float d = t32[tid * 32 + c] - m; var += d * d; }
    var *= (1.f / 32.f);
    const float inv = rsqrtf(var + 1e-5f);
    for (int c = 0; c < 32; ++c) {
      const float val = (t32[tid * 32 + c] - m) * inv * P.ln1w[c] + P.ln1b[c];
      h16[tid * 64 + 32 + c] = (_Float16)val;
      h16[tid * 64 + c] = (_Float16)x032[tid * 32 + c];
    }
  }
  for (int i = tid; i < 64 * 64; i += 128) Wc[i] = (_Float16)P.amlp0[i];
  __syncthreads();
  gemm_wmma(h16, 64, Wc, 64, nullptr, o16, 64, nullptr, 0, 64, 64, 64, ACT_RELU);
  __syncthreads();
  for (int i = tid; i < 32 * 64; i += 128) Wc[i] = (_Float16)P.amlp1[i];
  __syncthreads();
  gemm_wmma(o16, 64, Wc, 64, nullptr, nullptr, 0, t32, 32, 64, 32, 64, ACT_NONE);
  __syncthreads();
  if (tid < 64) {                              // ln2 + residual
    float m = 0.f;
    for (int c = 0; c < 32; ++c) m += t32[tid * 32 + c];
    m *= (1.f / 32.f);
    float var = 0.f;
    for (int c = 0; c < 32; ++c) { const float d = t32[tid * 32 + c] - m; var += d * d; }
    var *= (1.f / 32.f);
    const float inv = rsqrtf(var + 1e-5f);
    for (int c = 0; c < 32; ++c) {
      const float xo = x032[tid * 32 + c] +
                       (t32[tid * 32 + c] - m) * inv * P.ln2w[c] + P.ln2b[c];
      x16[tid * 32 + c] = (_Float16)xo;
    }
  }
  for (int i = tid; i < 16 * 32; i += 128) Wc[i] = (_Float16)P.og0w[i];
  for (int i = tid; i < 64; i += 128) biasC[i] = (i < 16) ? P.og0b[i] : 0.f;
  __syncthreads();
  gemm_wmma(x16, 32, Wc, 32, biasC, o16, 16, nullptr, 0, 64, 16, 32, ACT_ELU);
  __syncthreads();
  if (tid < 64) {                              // out_geo 16 -> 1, relu
    float a = P.og1b[0];
    for (int j = 0; j < 16; ++j) a += (float)o16[tid * 16 + j] * P.og1w[j];
    ws[OFF_SIGMA + (size_t)seq * SNN + tid] = fmaxf(a, 0.f);
  }
}

// ============ Kernel D1: softmax stats over RN axis (per b,v,s) ============
__global__ __launch_bounds__(256)
void k_softstats(float* __restrict__ ws) {
  __shared__ float sm[256];
  const int blk = blockIdx.x;                 // (b*NV+v)*SN + s
  const int s = blk % SNN;
  const int bv = blk / SNN;                   // b*NV+v
  const int tid = threadIdx.x;

  float mx = -3.4e38f;
  for (int k = 0; k < RNN / 256; ++k) {
    const int r = k * 256 + tid;
    mx = fmaxf(mx, ws[OFF_LOGIT + ((size_t)bv * RNN + r) * SNN + s]);
  }
  sm[tid] = mx;
  __syncthreads();
  for (int o = 128; o > 0; o >>= 1) {
    if (tid < o) sm[tid] = fmaxf(sm[tid], sm[tid + o]);
    __syncthreads();
  }
  const float gmax = sm[0];
  __syncthreads();
  float se = 0.f;
  for (int k = 0; k < RNN / 256; ++k) {
    const int r = k * 256 + tid;
    se += __expf(ws[OFF_LOGIT + ((size_t)bv * RNN + r) * SNN + s] - gmax);
  }
  sm[tid] = se;
  __syncthreads();
  for (int o = 128; o > 0; o >>= 1) {
    if (tid < o) sm[tid] += sm[tid + o];
    __syncthreads();
  }
  if (tid == 0) {
    ws[OFF_LMAX + blk] = gmax;
    ws[OFF_LSUM + blk] = sm[0];
  }
}

// ============ Kernel D2: blend rgb over views + sigma mask ============
__global__ __launch_bounds__(256)
void k_blend(const float* __restrict__ ws, float* __restrict__ out) {
  const int i = blockIdx.x * 256 + threadIdx.x;   // < NPS
  const int b = i / (RNN * SNN);
  const int r = (i / SNN) % RNN;
  const int s = i % SNN;
  float rgb[3] = {0.f, 0.f, 0.f};
  for (int v = 0; v < NVV; ++v) {
    const size_t gidx = ((size_t)(b * NVV + v) * RNN + r) * SNN + s;
    const int blk = (b * NVV + v) * SNN + s;
    const float p = __expf(ws[OFF_LOGIT + gidx] - ws[OFF_LMAX + blk]) / ws[OFF_LSUM + blk];
    for (int c = 0; c < 3; ++c) rgb[c] += ws[OFF_RGBIN + gidx * 3 + c] * p;
  }
  const size_t ps = (size_t)(b * RNN + r) * SNN + s;
  const float sig = (ws[OFF_NUMV + ps] < 1.f) ? 0.f : ws[OFF_SIGMA + ps];
  out[ps * 4 + 0] = rgb[0];
  out[ps * 4 + 1] = rgb[1];
  out[ps * 4 + 2] = rgb[2];
  out[ps * 4 + 3] = sig;
}

extern "C" void kernel_launch(void* const* d_in, const int* in_sizes, int n_in,
                              void* d_out, int out_size, void* d_ws, size_t ws_size,
                              hipStream_t stream) {
  (void)in_sizes; (void)out_size; (void)ws_size;
  if (n_in < 47) return;   // expects 6 tensors + 40 param leaves + scalar s

  const float* p3d   = (const float*)d_in[0];
  const float* simgs = (const float*)d_in[1];
  const float* rpi   = (const float*)d_in[2];
  const float* spi   = (const float*)d_in[3];
  const float* spose = (const float*)d_in[4];
  const float* sfeat = (const float*)d_in[5];

  Params P;
  P.rd0w   = (const float*)d_in[6];  P.rd0b   = (const float*)d_in[7];
  P.rd1w   = (const float*)d_in[8];  P.rd1b   = (const float*)d_in[9];
  P.base0w = (const float*)d_in[10]; P.base0b = (const float*)d_in[11];
  P.base1w = (const float*)d_in[12]; P.base1b = (const float*)d_in[13];
  P.vis0w  = (const float*)d_in[14]; P.vis0b  = (const float*)d_in[15];
  P.vis1w  = (const float*)d_in[16]; P.vis1b  = (const float*)d_in[17];
  P.v20w   = (const float*)d_in[18]; P.v20b   = (const float*)d_in[19];
  P.v21w   = (const float*)d_in[20]; P.v21b   = (const float*)d_in[21];
  P.geo0w  = (const float*)d_in[22]; P.geo0b  = (const float*)d_in[23];
  P.geo1w  = (const float*)d_in[24]; P.geo1b  = (const float*)d_in[25];
  P.og0w   = (const float*)d_in[26]; P.og0b   = (const float*)d_in[27];
  P.og1w   = (const float*)d_in[28]; P.og1b   = (const float*)d_in[29];
  P.rgb0w  = (const float*)d_in[30]; P.rgb0b  = (const float*)d_in[31];
  P.rgb1w  = (const float*)d_in[32]; P.rgb1b  = (const float*)d_in[33];
  P.rgb2w  = (const float*)d_in[34]; P.rgb2b  = (const float*)d_in[35];
  P.aq     = (const float*)d_in[36]; P.ak     = (const float*)d_in[37];
  P.av     = (const float*)d_in[38]; P.amerge = (const float*)d_in[39];
  P.amlp0  = (const float*)d_in[40]; P.amlp1  = (const float*)d_in[41];
  P.ln1w   = (const float*)d_in[42]; P.ln1b   = (const float*)d_in[43];
  P.ln2w   = (const float*)d_in[44]; P.ln2b   = (const float*)d_in[45];
  P.sparam = (const float*)d_in[46];

  float* ws = (float*)d_ws;
  float* out = (float*)d_out;

  (void)hipFuncSetAttribute((const void*)k_perray,
                            hipFuncAttributeMaxDynamicSharedMemorySize, SMEM_B);
  (void)hipFuncSetAttribute((const void*)k_attn,
                            hipFuncAttributeMaxDynamicSharedMemorySize, SMEM_C);

  k_project<<<NPT / 256, 256, 0, stream>>>(p3d, simgs, rpi, spi, spose, sfeat, P, ws, out);
  k_perray<<<BB * RNN, 256, SMEM_B, stream>>>(P, ws);
  k_attn<<<BB * RNN, 128, SMEM_C, stream>>>(P, ws);
  k_softstats<<<BB * NVV * SNN, 256, 0, stream>>>(ws);
  k_blend<<<NPS / 256, 256, 0, stream>>>(ws, out);
}
